// VaeTransformer_69088843923802
// MI455X (gfx1250) — compile-verified, matrix operand order checked by reading, not run on hardware
//
#include <hip/hip_runtime.h>

typedef __bf16 bf16;
typedef bf16  v16bf __attribute__((ext_vector_type(16)));
typedef float v8f   __attribute__((ext_vector_type(8)));

union FragU { v16bf v; uint4 q[2]; };

__device__ __forceinline__ void load_frag(const bf16* base, int row, int ld,
                                          int k0, int half, FragU& f) {
  const bf16* p = base + (size_t)row * (size_t)ld + k0 + half * 8;
  f.q[0] = *(const uint4*)(p);
  f.q[1] = *(const uint4*)(p + 16);
}

// TN GEMM: A (M,K) row-major bf16, Bt (N,K) row-major bf16, C (M,N).
// Block tile 128x128, 4 waves (wave32) in 2x2, each wave 64x64 via 4x4 wmma accums.
// Per K-step per wave: 8 fragment loads (16 x b128) feed 16 v_wmma ops.
template<int OUT_BF16, int HAS_BIAS, int RELU>
__global__ __launch_bounds__(128)
void k_gemm(const bf16* __restrict__ A, const bf16* __restrict__ Bt,
            void* __restrict__ Cv, const float* __restrict__ bias,
            int M, int N, int K, size_t sA, size_t sB, size_t sC)
{
  (void)M;
  const int z = blockIdx.z;
  A  += (size_t)z * sA;
  Bt += (size_t)z * sB;
  const int wave = threadIdx.x >> 5;        // 0..3
  const int lane = threadIdx.x & 31;
  const int wm = wave >> 1;                 // 0..1
  const int wn = wave & 1;                  // 0..1
  const int mW = blockIdx.x * 128 + wm * 64;
  const int nW = blockIdx.y * 128 + wn * 64;
  const int r    = lane & 15;
  const int half = lane >> 4;

  v8f acc[4][4];
  #pragma unroll
  for (int i = 0; i < 4; i++)
    #pragma unroll
    for (int j = 0; j < 4; j++)
      #pragma unroll
      for (int e = 0; e < 8; e++) acc[i][j][e] = 0.0f;

  for (int k0 = 0; k0 < K; k0 += 32) {
    FragU a[4], b[4];
    #pragma unroll
    for (int i = 0; i < 4; i++) load_frag(A,  mW + i*16 + r, K, k0, half, a[i]);
    #pragma unroll
    for (int j = 0; j < 4; j++) load_frag(Bt, nW + j*16 + r, K, k0, half, b[j]);
    if (k0 + 32 < K) {
      __builtin_prefetch(A  + (size_t)(mW + r) * (size_t)K + k0 + 32, 0, 1);
      __builtin_prefetch(Bt + (size_t)(nW + r) * (size_t)K + k0 + 32, 0, 1);
    }
    #pragma unroll
    for (int i = 0; i < 4; i++)
      #pragma unroll
      for (int j = 0; j < 4; j++)
        acc[i][j] = __builtin_amdgcn_wmma_f32_16x16x32_bf16(
            false, a[i].v, false, b[j].v, (short)0, acc[i][j], false, false);
  }

  float* Cf = (float*)Cv;
  bf16*  Ch = (bf16*)Cv;
  #pragma unroll
  for (int j = 0; j < 4; j++) {
    const int col = nW + j*16 + r;
    const float bv = HAS_BIAS ? bias[col] : 0.0f;
    #pragma unroll
    for (int i = 0; i < 4; i++) {
      #pragma unroll
      for (int e = 0; e < 8; e++) {
        const int row = mW + i*16 + half*8 + e;   // C layout: VGPR e -> M=e (+8 for hi lanes)
        float v0 = acc[i][j][e] + bv;
        if (RELU) v0 = v0 > 0.0f ? v0 : 0.0f;
        const size_t idx = (size_t)z * sC + (size_t)row * (size_t)N + col;
        if (OUT_BF16) Ch[idx] = (bf16)v0; else Cf[idx] = v0;
      }
    }
  }
}

// Transpose-convert f32 (K,N) -> bf16 (N,K)
__global__ __launch_bounds__(256)
void k_cvtT(const float* __restrict__ in, bf16* __restrict__ out, int K, int N) {
  size_t idx = (size_t)blockIdx.x * 256 + threadIdx.x;
  if (idx >= (size_t)K * (size_t)N) return;
  int n = (int)(idx / K), k = (int)(idx % K);
  out[idx] = (bf16)in[(size_t)k * (size_t)N + n];
}

__global__ __launch_bounds__(256)
void k_cvt(const float* __restrict__ in, bf16* __restrict__ out, size_t n) {
  size_t idx = (size_t)blockIdx.x * 256 + threadIdx.x;
  if (idx < n) out[idx] = (bf16)in[idx];
}

// emb_pos = concat(emb[x], PE); writes f32+bf16 copies for E (kept) and X (working).
__global__ __launch_bounds__(256)
void k_embed(const int* __restrict__ x, const float* __restrict__ emb,
             float* __restrict__ E, bf16* __restrict__ Eb,
             float* __restrict__ X, bf16* __restrict__ Xb) {
  const size_t row = blockIdx.x;        // 0..8191  (b*256 + t)
  const int c = threadIdx.x;            // handles c and 256+c
  const int t = (int)(row & 255);
  const int tok = x[row];
  const float v0 = emb[(size_t)tok * 256 + c];
  const int k = c >> 1;
  const float div = expf((float)(2 * k) * (-9.210340371976184f / 256.0f));
  const float arg = (float)t * div;
  const float v1 = (c & 1) ? cosf(arg) : sinf(arg);
  const size_t base = row * 512;
  E[base + c] = v0;       E[base + 256 + c] = v1;
  X[base + c] = v0;       X[base + 256 + c] = v1;
  Eb[base + c] = (bf16)v0; Eb[base + 256 + c] = (bf16)v1;
  Xb[base + c] = (bf16)v0; Xb[base + 256 + c] = (bf16)v1;
}

// softmax over one 256-wide row of one attention chunk; writes bf16 probs
__global__ __launch_bounds__(256)
void k_softmax(const float* __restrict__ S, bf16* __restrict__ Sb, int causal) {
  __shared__ float red[256];
  const int i = blockIdx.x;             // row in chunk
  const int z = blockIdx.y;             // chunk 0..255
  const int j = threadIdx.x;
  const float scale = 0.044194173824159216f;   // 1/sqrt(512)
  const size_t base = (size_t)z * 65536 + (size_t)i * 256;
  float v = S[base + j] * scale;
  if (causal && j > i) v = -__builtin_inff();
  red[j] = v; __syncthreads();
  for (int s = 128; s > 0; s >>= 1) { if (j < s) red[j] = fmaxf(red[j], red[j + s]); __syncthreads(); }
  const float mx = red[0]; __syncthreads();
  const float e = __expf(v - mx);
  red[j] = e; __syncthreads();
  for (int s = 128; s > 0; s >>= 1) { if (j < s) red[j] += red[j + s]; __syncthreads(); }
  Sb[base + j] = (bf16)(e / red[0]);
}

// X = LayerNorm(Xin + Y) * g + b  (row length 512); writes f32 and bf16
__global__ __launch_bounds__(256)
void k_add_ln(const float* Xin, const float* __restrict__ Yin,
              const float* __restrict__ g, const float* __restrict__ bb,
              float* Xout, bf16* __restrict__ Xbout) {
  __shared__ float red[256];
  const size_t base = (size_t)blockIdx.x * 512;
  const int t = threadIdx.x;
  const float s0 = Xin[base + t]       + Yin[base + t];
  const float s1 = Xin[base + t + 256] + Yin[base + t + 256];
  red[t] = s0 + s1; __syncthreads();
  for (int s = 128; s > 0; s >>= 1) { if (t < s) red[t] += red[t + s]; __syncthreads(); }
  const float mean = red[0] * (1.0f / 512.0f); __syncthreads();
  const float d0 = s0 - mean, d1 = s1 - mean;
  red[t] = d0 * d0 + d1 * d1; __syncthreads();
  for (int s = 128; s > 0; s >>= 1) { if (t < s) red[t] += red[t + s]; __syncthreads(); }
  const float inv = rsqrtf(red[0] * (1.0f / 512.0f) + 1e-5f);
  const float o0 = d0 * inv * g[t] + bb[t];
  const float o1 = d1 * inv * g[t + 256] + bb[t + 256];
  Xout[base + t] = o0;             Xout[base + t + 256] = o1;
  Xbout[base + t] = (bf16)o0;      Xbout[base + t + 256] = (bf16)o1;
}

// per-chunk transpose V (256x512) -> Vt (512x256), bf16
__global__ __launch_bounds__(256)
void k_transposeV(const bf16* __restrict__ V, bf16* __restrict__ Vt) {
  const size_t idx = (size_t)blockIdx.x * 256 + threadIdx.x; // 256 chunks * 131072
  const size_t z = idx >> 17;
  const int rem = (int)(idx & 131071);
  const int c = rem >> 8, j = rem & 255;
  Vt[idx] = V[z * 131072 + (size_t)j * 512 + c];
}

// hm[b,c] = mean over t of X[b,t,c]
__global__ void k_mean_t(const float* __restrict__ X, float* __restrict__ hm) {
  const int b = blockIdx.x, c = threadIdx.x;    // blockDim = 512
  float s = 0.0f;
  for (int t = 0; t < 256; t++) s += X[((size_t)b * 256 + t) * 512 + c];
  hm[b * 512 + c] = s * (1.0f / 256.0f);
}

// mu/logvar/z; mu & logvar go straight to d_out tail
__global__ void k_latent(const float* __restrict__ hm,
                         const float* __restrict__ muW, const float* __restrict__ mub,
                         const float* __restrict__ lvW, const float* __restrict__ lvb,
                         const float* __restrict__ eps,
                         float* __restrict__ muOut, float* __restrict__ lvOut,
                         bf16* __restrict__ zb) {
  const int b = blockIdx.x, l = threadIdx.x;    // blockDim = 128
  const float* h = hm + b * 512;
  float sm = 0.0f, sv = 0.0f;
  for (int k = 0; k < 512; k++) { const float hv = h[k]; sm += hv * muW[k * 128 + l]; sv += hv * lvW[k * 128 + l]; }
  const float mu = sm + mub[l], lv = sv + lvb[l];
  muOut[b * 128 + l] = mu;
  lvOut[b * 128 + l] = lv;
  zb[b * 128 + l] = (bf16)(mu + expf(lv) * eps[b * 128 + l]);
}

// decoder input: Ab (8192 x 640) = [Eb | broadcast z_b]
__global__ __launch_bounds__(256)
void k_decin(const bf16* __restrict__ Eb, const bf16* __restrict__ zb, bf16* __restrict__ Ab) {
  const size_t idx = (size_t)blockIdx.x * 256 + threadIdx.x;   // 8192*640
  const size_t row = idx / 640; const int col = (int)(idx % 640);
  Ab[idx] = (col < 512) ? Eb[row * 512 + col] : zb[(row >> 8) * 128 + (col - 512)];
}

static void run_block(const bf16* wq, const bf16* wk, const bf16* wv, const bf16* wo,
                      const bf16* w1, const bf16* w2,
                      const float* ln1g, const float* ln1b,
                      const float* b1, const float* b2,
                      const float* ln2g, const float* ln2b,
                      float* Xf, bf16* Xb, bf16* Q, bf16* Kb, bf16* V,
                      float* Sf, bf16* Sb, float* Yf, bf16* FH,
                      int causal, hipStream_t s)
{
  const dim3 gblk(128);   // GEMM blocks: 4 waves
  const dim3 blk(256);
  k_gemm<1,0,0><<<dim3(64,32,1),gblk,0,s>>>(Xb, wq, Q,  nullptr, 8192,4096,512, 0,0,0);
  k_gemm<1,0,0><<<dim3(64,32,1),gblk,0,s>>>(Xb, wk, Kb, nullptr, 8192,4096,512, 0,0,0);
  k_gemm<1,0,0><<<dim3(64,32,1),gblk,0,s>>>(Xb, wv, V,  nullptr, 8192,4096,512, 0,0,0);
  // logits per chunk: S = Q * K^T  (TN with row-major Q,K)
  k_gemm<0,0,0><<<dim3(2,2,256),gblk,0,s>>>(Q, Kb, Sf, nullptr, 256,256,512, 131072,131072,65536);
  k_softmax<<<dim3(256,256),blk,0,s>>>(Sf, Sb, causal);
  bf16* Vt = Q;                                   // Q dead after S
  k_transposeV<<<131072,blk,0,s>>>(V, Vt);
  bf16* O = Kb;                                   // K dead after S
  k_gemm<1,0,0><<<dim3(2,4,256),gblk,0,s>>>(Sb, Vt, O, nullptr, 256,512,256, 65536,131072,131072);
  k_gemm<0,0,0><<<dim3(64,4,1),gblk,0,s>>>(O, wo, Yf, nullptr, 8192,512,4096, 0,0,0);
  k_add_ln<<<8192,blk,0,s>>>(Xf, Yf, ln1g, ln1b, Xf, Xb);
  k_gemm<1,1,1><<<dim3(64,16,1),gblk,0,s>>>(Xb, w1, FH, b1, 8192,2048,512, 0,0,0);
  k_gemm<0,1,0><<<dim3(64,4,1),gblk,0,s>>>(FH, w2, Yf, b2, 8192,512,2048, 0,0,0);
  k_add_ln<<<8192,blk,0,s>>>(Xf, Yf, ln2g, ln2b, Xf, Xb);
}

extern "C" void kernel_launch(void* const* d_in, const int* in_sizes, int n_in,
                              void* d_out, int out_size, void* d_ws, size_t ws_size,
                              hipStream_t stream)
{
  (void)in_sizes; (void)n_in; (void)out_size;
  char* ws = (char*)d_ws;
  size_t off = 0;
  auto alloc = [&](size_t bytes) -> size_t {
    off = (off + 255) & ~(size_t)255; const size_t o = off; off += bytes; return o;
  };
  auto P = [&](int i) -> const float* { return (const float*)d_in[i]; };

  // --- workspace arena (bf16 transposed weights + activations) ---
  const size_t o_wq0 = alloc(4096*512*2);
  const size_t o_wk0 = alloc(4096*512*2);
  const size_t o_wv0 = alloc(4096*512*2);
  const size_t o_wo0 = alloc(512*4096*2);
  const size_t o_w10 = alloc(2048*512*2);
  const size_t o_w20 = alloc(512*2048*2);
  const size_t o_wq1 = alloc(4096*512*2);
  const size_t o_wk1 = alloc(4096*512*2);
  const size_t o_wv1 = alloc(4096*512*2);
  const size_t o_wo1 = alloc(512*4096*2);
  const size_t o_w11 = alloc(2048*512*2);
  const size_t o_w21 = alloc(512*2048*2);
  const size_t o_z2h = alloc(512*640*2);
  const size_t o_outw= alloc(128*512*2);
  const size_t o_E   = alloc((size_t)8192*512*4);
  const size_t o_Eb  = alloc((size_t)8192*512*2);
  const size_t o_X   = alloc((size_t)8192*512*4);
  const size_t o_Xb  = alloc((size_t)8192*512*2);
  const size_t o_Q   = alloc((size_t)8192*4096*2);
  const size_t o_K   = alloc((size_t)8192*4096*2);
  const size_t o_V   = alloc((size_t)8192*4096*2);
  const size_t o_Sf  = alloc((size_t)256*65536*4);
  const size_t o_Sb  = alloc((size_t)256*65536*2);
  const size_t o_Y   = alloc((size_t)8192*512*4);
  const size_t o_hm  = alloc((size_t)32*512*4);
  const size_t o_zb  = alloc((size_t)32*128*2);
  if (off > ws_size) return;   // workspace too small: cannot run

  float* Ef  = (float*)(ws + o_E);   bf16* Eb = (bf16*)(ws + o_Eb);
  float* Xf  = (float*)(ws + o_X);   bf16* Xb = (bf16*)(ws + o_Xb);
  bf16*  Q   = (bf16*)(ws + o_Q);
  bf16*  Kb  = (bf16*)(ws + o_K);
  bf16*  V   = (bf16*)(ws + o_V);
  float* Sf  = (float*)(ws + o_Sf);  bf16* Sb = (bf16*)(ws + o_Sb);
  float* Yf  = (float*)(ws + o_Y);
  bf16*  FH  = (bf16*)(ws + o_Sf);   // FF hidden aliases S (dead by then)
  float* hm  = (float*)(ws + o_hm);
  bf16*  zb  = (bf16*)(ws + o_zb);
  bf16*  Ab  = Q;                    // decoder concat input aliases Q

  const dim3 blk(256);
  const dim3 gblk(128);
  // --- convert + transpose weights to bf16 (N,K) ---
  struct CvtJob { const float* in; size_t out; int K, N; };
  const CvtJob jobs[14] = {
    { P(3),  o_wq0, 512, 4096 },   { P(4),  o_wk0, 512, 4096 }, { P(5),  o_wv0, 512, 4096 },
    { P(6),  o_wo0, 4096, 512 },   { P(9),  o_w10, 512, 2048 }, { P(11), o_w20, 2048, 512 },
    { P(15), o_wq1, 512, 4096 },   { P(16), o_wk1, 512, 4096 }, { P(17), o_wv1, 512, 4096 },
    { P(18), o_wo1, 4096, 512 },   { P(21), o_w11, 512, 2048 }, { P(23), o_w21, 2048, 512 },
    { P(31), o_z2h, 640, 512 },    { P(33), o_outw, 512, 128 },
  };
  for (int i = 0; i < 14; i++) {
    const size_t n = (size_t)jobs[i].K * jobs[i].N;
    k_cvtT<<<(unsigned)((n + 255) / 256), blk, 0, stream>>>(jobs[i].in, (bf16*)(ws + jobs[i].out),
                                                            jobs[i].K, jobs[i].N);
  }

  // --- embedding + positional encoding ---
  k_embed<<<8192, blk, 0, stream>>>((const int*)d_in[0], P(2), Ef, Eb, Xf, Xb);

  // --- encoder block (non-causal) ---
  run_block((bf16*)(ws+o_wq0), (bf16*)(ws+o_wk0), (bf16*)(ws+o_wv0), (bf16*)(ws+o_wo0),
            (bf16*)(ws+o_w10), (bf16*)(ws+o_w20),
            P(7), P(8), P(10), P(12), P(13), P(14),
            Xf, Xb, Q, Kb, V, Sf, Sb, Yf, FH, /*causal=*/0, stream);

  // --- latent head: mu/logvar -> d_out tail, z -> zb ---
  float* out_f = (float*)d_out;
  k_mean_t<<<32, 512, 0, stream>>>(Xf, hm);
  k_latent<<<32, 128, 0, stream>>>(hm, P(27), P(28), P(29), P(30), P(1),
                                   out_f + 1048576, out_f + 1048576 + 4096, zb);

  // --- decoder input: hd = [emb_pos | z] @ z2h + b ---
  k_decin<<<20480, blk, 0, stream>>>(Eb, zb, Ab);
  k_gemm<0,1,0><<<dim3(64,4,1), gblk, 0, stream>>>(Ab, (bf16*)(ws+o_z2h), Xf, P(32),
                                                   8192,512,640, 0,0,0);
  k_cvt<<<16384, blk, 0, stream>>>(Xf, Xb, (size_t)8192*512);

  // --- decoder block (causal) ---
  run_block((bf16*)(ws+o_wq1), (bf16*)(ws+o_wk1), (bf16*)(ws+o_wv1), (bf16*)(ws+o_wo1),
            (bf16*)(ws+o_w11), (bf16*)(ws+o_w21),
            P(19), P(20), P(22), P(24), P(25), P(26),
            Xf, Xb, Q, Kb, V, Sf, Sb, Yf, FH, /*causal=*/1, stream);

  // --- output projection -> logits at d_out[0] ---
  k_gemm<0,1,0><<<dim3(64,1,1), gblk, 0, stream>>>(Xb, (bf16*)(ws+o_outw), out_f, P(34),
                                                   8192,128,512, 0,0,0);
}
